// EnhancedHybridFakeNewsClassifier_71605694759284
// MI455X (gfx1250) — compile-verified, matrix operand order checked by reading, not run on hardware
//
#include <hip/hip_runtime.h>
#include <hip/hip_bf16.h>
#include <math.h>

// ---------------- model dims ----------------
#define N_NODES 50000
#define N_EDGES 1600000
#define F_IN    300
#define E_DIM   256
#define BERT_D  768
#define BATCH   1024
#define SEQ_L   64
#define HEADS   4
#define DHEAD   64
#define EPSV    1e-5f

typedef __attribute__((ext_vector_type(16))) _Float16 v16h;
typedef __attribute__((ext_vector_type(8)))  float    v8f;
typedef __attribute__((ext_vector_type(4)))  unsigned u32x4;
typedef __attribute__((ext_vector_type(8)))  int      i32x8;
typedef __attribute__((ext_vector_type(4)))  int      i32x4;

#ifndef __has_builtin
#define __has_builtin(x) 0
#endif
#if __has_builtin(__builtin_amdgcn_tensor_load_to_lds)
#define HAVE_TDM 1
#else
#define HAVE_TDM 0
#endif

// K index base for 16-bit A/B WMMA fragment layout (ISA 7.12.2):
// lanes 0-15 hold K={0..7,16..23} pairs in VGPRs 0..7, lanes 16-31 the +8 halves.
__device__ __forceinline__ int kbase_of(int v, int hi) {
  return (v < 4) ? (2 * v + 8 * hi) : (16 + 2 * (v - 4) + 8 * hi);
}

// One async 16B global->LDS copy (ASYNCcnt-tracked, ISA §15.18.3 op 98).
__device__ __forceinline__ void async_copy_b128(const float* g, void* lds) {
  unsigned lo = (unsigned)(size_t)lds;  // LDS aperture: addr[31:0] is the LDS offset
  asm volatile("global_load_async_to_lds_b128 %0, %1, off"
               :: "v"(lo), "v"(g) : "memory");
}
__device__ __forceinline__ void wait_async_all() {
  asm volatile("s_wait_asynccnt 0x0" ::: "memory");
}
__device__ __forceinline__ void wait_tensor_all() {
#if __has_builtin(__builtin_amdgcn_s_wait_tensorcnt)
  __builtin_amdgcn_s_wait_tensorcnt(0);
#else
  asm volatile("s_wait_tensorcnt 0x0" ::: "memory");
#endif
}

#if HAVE_TDM
// Issue one TDM 2D-tile load (D# groups 0/1 per ISA §8.3/§8.4; groups 2/3 zero).
// rows x rowlen tile of f32 at gp (row stride = stride elems) -> contiguous LDS
// rows of rowlen floats at lds_off. OOB (tensor_dim) reads return zero.
__device__ __forceinline__ void tdm_load_tile(const float* gp, unsigned lds_off,
                                              unsigned rowlen_tile, unsigned rows_tile,
                                              unsigned rowlen_tensor, unsigned rows_tensor,
                                              unsigned stride) {
  unsigned long long ga = (unsigned long long)(size_t)gp;
  u32x4 g0;
  g0[0] = 1u;                                        // count=1 valid user descriptor
  g0[1] = lds_off;                                   // lds_addr (bytes)
  g0[2] = (unsigned)(ga & 0xFFFFFFFFull);            // global_addr[31:0]
  g0[3] = (unsigned)((ga >> 32) & 0x1FFFFFFull)      // global_addr[56:32]
          | (2u << 30);                              // type=2 ("image")
  i32x8 g1;
  g1[0] = (int)(2u << 16);                           // data_size=2 -> 4-byte elems
  g1[1] = (int)((rowlen_tensor & 0xFFFFu) << 16);    // tensor_dim0[15:0]
  g1[2] = (int)(((rowlen_tensor >> 16) & 0xFFFFu)    // tensor_dim0[31:16]
          | ((rows_tensor & 0xFFFFu) << 16));        // tensor_dim1[15:0]
  g1[3] = (int)(((rows_tensor >> 16) & 0xFFFFu)      // tensor_dim1[31:16]
          | (rowlen_tile << 16));                    // tile_dim0
  g1[4] = (int)rows_tile;                            // tile_dim1 (tile_dim2=0)
  g1[5] = (int)stride;                               // tensor_dim0_stride (elems)
  g1[6] = 0;                                         // stride hi / dim1_stride (2D)
  g1[7] = 0;
  i32x4 z4 = {0, 0, 0, 0};
#if __clang_major__ >= 23
  i32x8 z8 = {0, 0, 0, 0, 0, 0, 0, 0};
  __builtin_amdgcn_tensor_load_to_lds(g0, g1, z4, z4, z8, 0);
#else
  __builtin_amdgcn_tensor_load_to_lds(g0, g1, z4, z4, 0);
#endif
}
#endif

// ---------------- tiled WMMA GEMM ----------------
// C = act(A[MxK] @ B + bias). Macro tile 64(M) x 128(N), BK=32 K-chunks.
// A staged via async global->LDS b128 copies; B staged via one TDM descriptor
// per chunk (wave 0). Double-buffered. 8 waves (2x4), each wave a 32x32
// register tile = 2x2 WMMA tiles. N % 128 == 0, K % 4 == 0 at all call sites.
#define BM 64
#define BN 128
#define BK 32

template <bool TRANSB, int ACT>
__global__ void __launch_bounds__(256)
wmma_gemm(const float* __restrict__ A, const float* __restrict__ B,
          const float* __restrict__ bias, float* __restrict__ C,
          int M, int N, int K, int lda, int ldb, int ldc) {
  __shared__ __align__(16) float sA[2][BM * BK];  // [m][k]
  __shared__ __align__(16) float sB[2][BK * BN];  // TRANSB ? [n][k] : [k][n]
  int tid = threadIdx.x;
  int wave = tid >> 5, lane = tid & 31, hi = lane >> 4, lr = lane & 15;
  int waveM = wave >> 2, waveN = wave & 3;
  int tileM0 = blockIdx.y * BM;
  int tileN0 = blockIdx.x * BN;

  // stage one K-chunk (kb) of A and B into LDS buffer `buf` (issue only)
  auto stage = [&](int buf, int kb) {
#pragma unroll
    for (int i = 0; i < 2; ++i) {        // A: 64x32 f32 = 512 x b128 (async)
      int c = tid + i * 256;
      int row = c >> 3;
      int col = (c & 7) * 4;
      int gm = tileM0 + row; if (gm >= M) gm = M - 1;        // clamp, masked later
      int gk = kb + col;     if (gk > K - 4) gk = K - 4;
      async_copy_b128(A + (size_t)gm * lda + gk, &sA[buf][row * BK + col]);
    }
#if HAVE_TDM
    if (wave == 0) {                     // B: one 2D TDM tile per chunk
      unsigned lo = (unsigned)(size_t)&sB[buf][0];
      if (TRANSB) {   // B[N][K]: 128 rows x 32 elems, remaining extent from tile
        tdm_load_tile(B + (size_t)tileN0 * ldb + kb, lo,
                      BK, BN, (unsigned)(K - kb), (unsigned)(N - tileN0),
                      (unsigned)ldb);
      } else {        // B[K][N]: 32 rows x 128 elems
        tdm_load_tile(B + (size_t)kb * ldb + tileN0, lo,
                      BN, BK, (unsigned)(N - tileN0), (unsigned)(K - kb),
                      (unsigned)ldb);
      }
    }
#else
#pragma unroll
    for (int i = 0; i < 4; ++i) {        // fallback: async b128 staging for B
      int c = tid + i * 256;
      if (TRANSB) {
        int nn = c >> 3;
        int kk = (c & 7) * 4;
        int gk = kb + kk; if (gk > K - 4) gk = K - 4;
        async_copy_b128(B + (size_t)(tileN0 + nn) * ldb + gk,
                        &sB[buf][nn * BK + kk]);
      } else {
        int kk = c >> 5;
        int nn = (c & 31) * 4;
        int gk = kb + kk; if (gk > K - 1) gk = K - 1;
        async_copy_b128(B + (size_t)gk * ldb + tileN0 + nn,
                        &sB[buf][kk * BN + nn]);
      }
    }
#endif
  };

  v8f acc00 = {}, acc01 = {}, acc10 = {}, acc11 = {};

  // fragment builders (f32 in LDS -> f16 frags); kv = valid K in this chunk
  auto frag_a = [&](int buf, int mt, int kv) {
    v16h a;
#pragma unroll
    for (int v = 0; v < 8; ++v) {
      int k0 = kbase_of(v, hi);
      float x0 = sA[buf][(mt + lr) * BK + k0];
      float x1 = sA[buf][(mt + lr) * BK + k0 + 1];
      if (k0     >= kv) x0 = 0.f;
      if (k0 + 1 >= kv) x1 = 0.f;
      a[2 * v] = (_Float16)x0; a[2 * v + 1] = (_Float16)x1;
    }
    return a;
  };
  auto frag_b = [&](int buf, int nt, int kv) {
    v16h b;
#pragma unroll
    for (int v = 0; v < 8; ++v) {
      int k0 = kbase_of(v, hi);
      float x0, x1;
      if (TRANSB) {
        x0 = sB[buf][(nt + lr) * BK + k0];
        x1 = sB[buf][(nt + lr) * BK + k0 + 1];
      } else {
        x0 = sB[buf][k0 * BN + nt + lr];
        x1 = sB[buf][(k0 + 1) * BN + nt + lr];
      }
      if (k0     >= kv) x0 = 0.f;
      if (k0 + 1 >= kv) x1 = 0.f;
      b[2 * v] = (_Float16)x0; b[2 * v + 1] = (_Float16)x1;
    }
    return b;
  };
  auto mma_step = [&](int buf, int kv) {
    v16h a0 = frag_a(buf, waveM * 32,      kv);
    v16h a1 = frag_a(buf, waveM * 32 + 16, kv);
    v16h b0 = frag_b(buf, waveN * 32,      kv);
    v16h b1 = frag_b(buf, waveN * 32 + 16, kv);
    acc00 = __builtin_amdgcn_wmma_f32_16x16x32_f16(false, a0, false, b0, (short)0, acc00, false, false);
    acc01 = __builtin_amdgcn_wmma_f32_16x16x32_f16(false, a0, false, b1, (short)0, acc01, false, false);
    acc10 = __builtin_amdgcn_wmma_f32_16x16x32_f16(false, a1, false, b0, (short)0, acc10, false, false);
    acc11 = __builtin_amdgcn_wmma_f32_16x16x32_f16(false, a1, false, b1, (short)0, acc11, false, false);
  };

  int nchunks = (K + BK - 1) / BK;
  int cur = 0;
  stage(0, 0);
  for (int ch = 0; ch < nchunks; ++ch) {
    wait_async_all();              // our wave's A copies for `cur` resident
#if HAVE_TDM
    if (wave == 0) wait_tensor_all();  // B tile resident (TENSORcnt is per-wave)
#endif
    __syncthreads();               // all waves' data + previous compute done
    if (ch + 1 < nchunks) stage(cur ^ 1, (ch + 1) * BK);  // prefetch next
    int kv = K - ch * BK;
    if (kv >= BK) mma_step(cur, BK);   // literal -> masking folds away
    else          mma_step(cur, kv);   // uniform branch: EXEC stays all-ones
    cur ^= 1;
  }

  // epilogue
  int n0 = tileN0 + waveN * 32;
#pragma unroll
  for (int tn = 0; tn < 2; ++tn) {
    int n = n0 + tn * 16 + lr;
    float bv = bias ? bias[n] : 0.f;
#pragma unroll
    for (int tm = 0; tm < 2; ++tm) {
      const v8f& acc = (tm == 0) ? (tn == 0 ? acc00 : acc01)
                                 : (tn == 0 ? acc10 : acc11);
#pragma unroll
      for (int v = 0; v < 8; ++v) {
        int mr = tileM0 + waveM * 32 + tm * 16 + v + 8 * hi;
        if (mr < M) {
          float val = acc[v] + bv;
          if (ACT == 1) val = fmaxf(val, 0.f);
          C[(size_t)mr * ldc + n] = val;
        }
      }
    }
  }
}

// ---------------- GCN helper kernels ----------------
__global__ void k_init_deg(float* deg) {
  int i = blockIdx.x * blockDim.x + threadIdx.x;
  if (i < N_NODES) deg[i] = 1.0f;  // segment_sum(ones) + 1.0
}

__global__ void k_deg_accum(const int* __restrict__ dst, float* __restrict__ deg) {
  int e = blockIdx.x * blockDim.x + threadIdx.x;
  if (e < N_EDGES) atomicAdd(&deg[dst[e]], 1.0f);
}

__global__ void k_dinv(float* deg) {
  int i = blockIdx.x * blockDim.x + threadIdx.x;
  if (i < N_NODES) deg[i] = rsqrtf(deg[i]);  // in place: deg -> dinv
}

__global__ void k_selfterm(const float* __restrict__ h, const float* __restrict__ dinv,
                           float* __restrict__ agg) {
  long long idx = (long long)blockIdx.x * blockDim.x + threadIdx.x;
  if (idx >= (long long)N_NODES * E_DIM) return;
  int node = (int)(idx >> 8);
  float di = dinv[node];
  agg[idx] = h[idx] * di * di;
}

// scatter-add over edges: agg[dst] += h[src] * dinv[src]*dinv[dst]
__global__ void k_edge_agg(const float* __restrict__ h, const int* __restrict__ src,
                           const int* __restrict__ dst, const float* __restrict__ dinv,
                           float* __restrict__ agg) {
  long long idx = (long long)blockIdx.x * blockDim.x + threadIdx.x;
  if (idx >= (long long)N_EDGES * (E_DIM / 4)) return;
  int e = (int)(idx >> 6);
  int f = (int)(idx & 63) * 4;
  int s = src[e], d = dst[e];
  float c = dinv[s] * dinv[d];
  const float* hp = h + (size_t)s * E_DIM + f;
  float* ap = agg + (size_t)d * E_DIM + f;
#pragma unroll
  for (int j = 0; j < 4; ++j) atomicAdd(ap + j, hp[j] * c);
}

__global__ void k_bias_relu(const float* __restrict__ agg, const float* __restrict__ bias,
                            float* __restrict__ out) {
  long long idx = (long long)blockIdx.x * blockDim.x + threadIdx.x;
  if (idx >= (long long)N_NODES * E_DIM) return;
  int f = (int)(idx & (E_DIM - 1));
  out[idx] = fmaxf(agg[idx] + bias[f], 0.f);
}

__global__ void k_gather(const float* __restrict__ gnn, const int* __restrict__ emap,
                         float* __restrict__ X) {
  long long idx = (long long)blockIdx.x * blockDim.x + threadIdx.x;
  if (idx >= (long long)BATCH * SEQ_L * E_DIM) return;
  int f = (int)(idx & (E_DIM - 1));
  int bl = (int)(idx >> 8);
  int id = emap[bl];
  if (id < 0) id = 0;  // ids_safe semantics
  X[idx] = gnn[(size_t)id * E_DIM + f];
}

// ---------------- attention: one block per (batch, head) ----------------
__global__ void __launch_bounds__(128)
k_attention(const float* __restrict__ qkv, const int* __restrict__ emap,
            float* __restrict__ ctx) {
  __shared__ _Float16 sq[SEQ_L * DHEAD];  // q (scaled), later reused for attn probs
  __shared__ _Float16 sk[SEQ_L * DHEAD];
  __shared__ _Float16 sv[SEQ_L * DHEAD];
  __shared__ float ss[SEQ_L * SEQ_L];
  __shared__ float svalid[SEQ_L];
  int b = blockIdx.x / HEADS;
  int h = blockIdx.x % HEADS;
  int tid = threadIdx.x;
  const float scale = 0.125f;  // 1/sqrt(DH=64)
  if (tid < SEQ_L) svalid[tid] = (emap[b * SEQ_L + tid] != -1) ? 1.f : 0.f;
  for (int i = tid; i < SEQ_L * DHEAD; i += 128) {
    int l = i >> 6, d = i & 63;
    const float* base = qkv + (size_t)(b * SEQ_L + l) * (3 * E_DIM) + h * DHEAD + d;
    sq[i] = (_Float16)(base[0] * scale);
    sk[i] = (_Float16)(base[E_DIM]);
    sv[i] = (_Float16)(base[2 * E_DIM]);
  }
  __syncthreads();
  int wave = tid >> 5, lane = tid & 31, hi = lane >> 4, lr = lane & 15;
  int tm = wave * 16;
  // scores = (q*scale) @ k^T, masked on key dim
  for (int nt = 0; nt < 4; ++nt) {
    v8f acc = {};
    for (int kb = 0; kb < DHEAD; kb += 32) {
      v16h af, bf;
#pragma unroll
      for (int v = 0; v < 8; ++v) {
        int k0 = kb + kbase_of(v, hi);
        af[2 * v]     = sq[(tm + lr) * DHEAD + k0];
        af[2 * v + 1] = sq[(tm + lr) * DHEAD + k0 + 1];
        bf[2 * v]     = sk[(nt * 16 + lr) * DHEAD + k0];  // k^T: B[kk][n]=k[n][kk]
        bf[2 * v + 1] = sk[(nt * 16 + lr) * DHEAD + k0 + 1];
      }
      acc = __builtin_amdgcn_wmma_f32_16x16x32_f16(false, af, false, bf,
                                                   (short)0, acc, false, false);
    }
#pragma unroll
    for (int v = 0; v < 8; ++v) {
      int mr = tm + v + 8 * hi;
      int nc = nt * 16 + lr;
      ss[mr * SEQ_L + nc] = (svalid[nc] > 0.f) ? acc[v] : -3.0e38f;
    }
  }
  __syncthreads();
  // row softmax; probs -> f16 in sq (reuse)
  if (tid < SEQ_L) {
    float mx = -3.0e38f;
    for (int j = 0; j < SEQ_L; ++j) mx = fmaxf(mx, ss[tid * SEQ_L + j]);
    float sum = 0.f;
    for (int j = 0; j < SEQ_L; ++j) {
      float e = __expf(ss[tid * SEQ_L + j] - mx);
      ss[tid * SEQ_L + j] = e;
      sum += e;
    }
    float inv = 1.f / sum;
    for (int j = 0; j < SEQ_L; ++j)
      sq[tid * SEQ_L + j] = (_Float16)(ss[tid * SEQ_L + j] * inv);
  }
  __syncthreads();
  // ctx = attn @ v, written in (B, L, H*DH) layout (matches transpose+reshape)
  for (int nt = 0; nt < 4; ++nt) {
    v8f acc = {};
    for (int kb = 0; kb < SEQ_L; kb += 32) {
      v16h af, bf;
#pragma unroll
      for (int v = 0; v < 8; ++v) {
        int k0 = kb + kbase_of(v, hi);
        af[2 * v]     = sq[(tm + lr) * SEQ_L + k0];
        af[2 * v + 1] = sq[(tm + lr) * SEQ_L + k0 + 1];
        bf[2 * v]     = sv[k0 * DHEAD + nt * 16 + lr];
        bf[2 * v + 1] = sv[(k0 + 1) * DHEAD + nt * 16 + lr];
      }
      acc = __builtin_amdgcn_wmma_f32_16x16x32_f16(false, af, false, bf,
                                                   (short)0, acc, false, false);
    }
#pragma unroll
    for (int v = 0; v < 8; ++v) {
      int mr = tm + v + 8 * hi;
      int nc = nt * 16 + lr;
      ctx[(size_t)(b * SEQ_L + mr) * E_DIM + h * DHEAD + nc] = acc[v];
    }
  }
}

// masked mean / single / zero selection
__global__ void k_gnn_emb(const float* __restrict__ att, const float* __restrict__ X,
                          const int* __restrict__ emap, float* __restrict__ out) {
  int b = blockIdx.x, f = threadIdx.x;
  float sum = 0.f;
  int cnt = 0;
  for (int l = 0; l < SEQ_L; ++l) {
    if (emap[b * SEQ_L + l] != -1) {
      sum += att[(size_t)(b * SEQ_L + l) * E_DIM + f];
      cnt++;
    }
  }
  float r;
  if (cnt > 1)       r = sum / (float)cnt;
  else if (cnt == 1) r = X[(size_t)(b * SEQ_L) * E_DIM + f];
  else               r = 0.f;
  out[(size_t)b * E_DIM + f] = r;
}

// per-column batchnorm over BATCH rows; writes into comb with stride ldo
__global__ void k_batchnorm(const float* __restrict__ x, const float* __restrict__ gamma,
                            const float* __restrict__ beta, float* __restrict__ out,
                            int cols, int ldo) {
  int c = blockIdx.x * blockDim.x + threadIdx.x;
  if (c >= cols) return;
  float s = 0.f, s2 = 0.f;
  for (int r = 0; r < BATCH; ++r) {
    float v = x[(size_t)r * cols + c];
    s += v;
    s2 += v * v;
  }
  float mu = s * (1.f / BATCH);
  float var = s2 * (1.f / BATCH) - mu * mu;
  float isd = rsqrtf(var + EPSV);
  float g = gamma[c], bt = beta[c];
  for (int r = 0; r < BATCH; ++r)
    out[(size_t)r * ldo + c] = (x[(size_t)r * cols + c] - mu) * isd * g + bt;
}

__global__ void k_fc3_sigmoid(const float* __restrict__ x2, const float* __restrict__ w,
                              const float* __restrict__ bias, float* __restrict__ out) {
  int b = blockIdx.x * blockDim.x + threadIdx.x;
  if (b >= BATCH) return;
  float s = bias[0];
  for (int j = 0; j < 256; ++j) s += x2[(size_t)b * 256 + j] * w[j];
  out[b] = 1.f / (1.f + __expf(-s));
}

// ---------------- launch ----------------
extern "C" void kernel_launch(void* const* d_in, const int* in_sizes, int n_in,
                              void* d_out, int out_size, void* d_ws, size_t ws_size,
                              hipStream_t stream) {
  const float* bert     = (const float*)d_in[0];
  const float* node_x   = (const float*)d_in[1];
  const int*   edge     = (const int*)d_in[2];
  const int*   emap     = (const int*)d_in[3];
  const float* conv1_w  = (const float*)d_in[4];
  const float* conv1_b  = (const float*)d_in[5];
  const float* conv2_w  = (const float*)d_in[6];
  const float* conv2_b  = (const float*)d_in[7];
  const float* inproj_w = (const float*)d_in[8];
  const float* inproj_b = (const float*)d_in[9];
  const float* outp_w   = (const float*)d_in[10];
  const float* outp_b   = (const float*)d_in[11];
  const float* bgam     = (const float*)d_in[12];
  const float* bbet     = (const float*)d_in[13];
  const float* ggam     = (const float*)d_in[14];
  const float* gbet     = (const float*)d_in[15];
  const float* fc1_w    = (const float*)d_in[16];
  const float* fc1_b    = (const float*)d_in[17];
  const float* fc2_w    = (const float*)d_in[18];
  const float* fc2_b    = (const float*)d_in[19];
  const float* fc3_w    = (const float*)d_in[20];
  const float* fc3_b    = (const float*)d_in[21];
  float* out = (float*)d_out;

  const int* esrc = edge;
  const int* edst = edge + N_EDGES;

  float* ws = (float*)d_ws;
  size_t o = 0;
  float* deg  = ws + o; o += 50176;                              // deg -> dinv in place
  float* h    = ws + o; o += (size_t)N_NODES * E_DIM;            // conv1 lin / conv1 out
  float* agg  = ws + o; o += (size_t)N_NODES * E_DIM;            // aggregation scratch
  float* gnn  = ws + o; o += (size_t)N_NODES * E_DIM;            // conv2 lin / final gnn
  float* X    = ws + o; o += (size_t)BATCH * SEQ_L * E_DIM;
  float* qkv  = ws + o; o += (size_t)BATCH * SEQ_L * 3 * E_DIM;
  float* ctx  = ws + o; o += (size_t)BATCH * SEQ_L * E_DIM;
  float* att  = ws + o; o += (size_t)BATCH * SEQ_L * E_DIM;
  float* gemb = ws + o; o += (size_t)BATCH * E_DIM;
  float* comb = ws + o; o += (size_t)BATCH * (BERT_D + E_DIM);
  float* x1   = ws + o; o += (size_t)BATCH * 512;
  float* x2   = ws + o; o += (size_t)BATCH * 256;
  (void)ws_size; (void)in_sizes; (void)n_in; (void)out_size;

  const int BL = BATCH * SEQ_L;  // 65536

  // degree / normalization
  k_init_deg<<<(N_NODES + 255) / 256, 256, 0, stream>>>(deg);
  k_deg_accum<<<(N_EDGES + 255) / 256, 256, 0, stream>>>(edst, deg);
  k_dinv<<<(N_NODES + 255) / 256, 256, 0, stream>>>(deg);

  long long nodef = (long long)N_NODES * E_DIM;
  long long edgef = (long long)N_EDGES * (E_DIM / 4);

  // conv1: h = node_x @ conv1_w ; agg = norm-sum ; h = relu(agg + b)
  wmma_gemm<false, 0><<<dim3(E_DIM / BN, (N_NODES + BM - 1) / BM), 256, 0, stream>>>(
      node_x, conv1_w, nullptr, h, N_NODES, E_DIM, F_IN, F_IN, E_DIM, E_DIM);
  k_selfterm<<<(int)((nodef + 255) / 256), 256, 0, stream>>>(h, deg, agg);
  k_edge_agg<<<(int)((edgef + 255) / 256), 256, 0, stream>>>(h, esrc, edst, deg, agg);
  k_bias_relu<<<(int)((nodef + 255) / 256), 256, 0, stream>>>(agg, conv1_b, h);

  // conv2: gnn = h @ conv2_w ; agg = norm-sum ; gnn = relu(agg + b)
  wmma_gemm<false, 0><<<dim3(E_DIM / BN, (N_NODES + BM - 1) / BM), 256, 0, stream>>>(
      h, conv2_w, nullptr, gnn, N_NODES, E_DIM, E_DIM, E_DIM, E_DIM, E_DIM);
  k_selfterm<<<(int)((nodef + 255) / 256), 256, 0, stream>>>(gnn, deg, agg);
  k_edge_agg<<<(int)((edgef + 255) / 256), 256, 0, stream>>>(gnn, esrc, edst, deg, agg);
  k_bias_relu<<<(int)((nodef + 255) / 256), 256, 0, stream>>>(agg, conv2_b, gnn);

  // gather + attention
  long long xf = (long long)BL * E_DIM;
  k_gather<<<(int)((xf + 255) / 256), 256, 0, stream>>>(gnn, emap, X);
  wmma_gemm<true, 0><<<dim3((3 * E_DIM) / BN, BL / BM), 256, 0, stream>>>(
      X, inproj_w, inproj_b, qkv, BL, 3 * E_DIM, E_DIM, E_DIM, E_DIM, 3 * E_DIM);
  k_attention<<<BATCH * HEADS, 128, 0, stream>>>(qkv, emap, ctx);
  wmma_gemm<true, 0><<<dim3(E_DIM / BN, BL / BM), 256, 0, stream>>>(
      ctx, outp_w, outp_b, att, BL, E_DIM, E_DIM, E_DIM, E_DIM, E_DIM);
  k_gnn_emb<<<BATCH, E_DIM, 0, stream>>>(att, X, emap, gemb);

  // batchnorms write straight into concat buffer
  k_batchnorm<<<(BERT_D + 255) / 256, 256, 0, stream>>>(bert, bgam, bbet, comb, BERT_D,
                                                        BERT_D + E_DIM);
  k_batchnorm<<<1, 256, 0, stream>>>(gemb, ggam, gbet, comb + BERT_D, E_DIM,
                                     BERT_D + E_DIM);

  // MLP head
  wmma_gemm<true, 1><<<dim3(512 / BN, BATCH / BM), 256, 0, stream>>>(
      comb, fc1_w, fc1_b, x1, BATCH, 512, BERT_D + E_DIM, BERT_D + E_DIM,
      BERT_D + E_DIM, 512);
  wmma_gemm<true, 1><<<dim3(256 / BN, BATCH / BM), 256, 0, stream>>>(
      x1, fc2_w, fc2_b, x2, BATCH, 256, 512, 512, 512, 256);
  k_fc3_sigmoid<<<(BATCH + 255) / 256, 256, 0, stream>>>(x2, fc3_w, fc3_b, out);
}
